// NeuralScene_51118700757189
// MI455X (gfx1250) — matrix operand order repaired and checked by reading, not compiled, not run on hardware
//
#include <hip/hip_runtime.h>
#include <hip/hip_bf16.h>
#include <cstdint>
#include <cstddef>

// ---------------- types ----------------
typedef __attribute__((ext_vector_type(16))) _Float16 v16h;
typedef __attribute__((ext_vector_type(8)))  _Float16 v8h;
typedef __attribute__((ext_vector_type(8)))  float    v8f;
typedef _Float16 half_t;

// ---------------- model dims ----------------
constexpr int NOBJ  = 6;     // 5 objects + dummy slot
constexpr int LFREQ = 10;
constexpr int DENC  = 60;    // 2*L*3
constexpr int MATD  = 48;
constexpr int HOBJ  = 64;
constexpr int HINT  = 128;
constexpr int ZK    = 56;    // MAT + GEO

constexpr int WAVES   = 16;            // waves per workgroup (4 per SIMD32)
constexpr int THREADS = WAVES * 32;

// ---------------- LDS layout (halves) ----------------
constexpr int FRH   = 512;                 // halves per fragment tile (32 lanes * 16)
constexpr int WO1_H = NOBJ * 2 * 4 * FRH;  // K=60->2 tiles, N=64->4 tiles
constexpr int WO2_H = NOBJ * 2 * 4 * FRH;
constexpr int WO3_H = NOBJ * 2 * 3 * FRH;  // N=48->3 tiles
constexpr int WI1_H = 2 * 8 * FRH;         // K=56->2, N=128->8
constexpr int WI2_H = 4 * 8 * FRH;         // K=128->4, N=128->8
constexpr int WI3_H = 4 * 1 * FRH;         // K=128->4, N=8->1 (padded)
// Per-wave staging: act = 4 K-tiles (interaction width); enc overlays act tiles 2..3
// (object layers only write tiles 0..1; enc is dead once interaction L1 writes act).
constexpr int ACT_H = WAVES * 4 * FRH;
constexpr int Z_H   = WAVES * 2 * FRH;
constexpr int TOT_H = WO1_H + WO2_H + WO3_H + WI1_H + WI2_H + WI3_H + ACT_H + Z_H;
constexpr int BIAS_F = 6*64 + 6*64 + 6*48 + 128 + 128 + 8;   // 1320 floats
constexpr int SMEM_BYTES = TOT_H * 2 + BIAS_F * 4 + WAVES * 16 * 4;  // + objid ints

// ---------------- A/B fragment addressing ----------------
// 16-bit A fragment (16x32, MxK) ISA layout:
//   lane = m + 16*hi ; e<8 -> k = e + 8*hi ; e>=8 -> k = 16 + (e-8) + 8*hi
// Activations are STORED directly in this layout so every fragment load is 2x b128.
__device__ __forceinline__ constexpr int frag_off(int m, int ng) {
  const int kt  = ng >> 5;
  const int k2  = ng & 31;
  const int hi2 = (k2 >> 3) & 1;
  const int e   = (k2 & 7) + ((k2 & 16) ? 8 : 0);
  return ((kt << 5) + (hi2 << 4) + m) * 16 + e;
}

// Load one 512-half fragment tile (A or pre-swizzled B): 2x ds_load_b128 per lane
__device__ __forceinline__ v16h load_frag(const half_t* base, int frag, int lane) {
  v16h v;
  __builtin_memcpy(&v, base + (size_t)((frag << 5) + lane) * 16, sizeof(v16h));
  return v;
}

// ---------------- generic GEMM layer ----------------
// D(16 x Np) = act(16 x Kp) * W + bias ; ACT: 0=relu 1=sigmoid 2=none
template <int KT, int NT, int ACT>
__device__ __forceinline__ void gemm_layer(const half_t* actIn, const half_t* wfrag,
                                           const float* bias, half_t* actOut,
                                           int Nvalid, int lane, unsigned rowmask) {
  const int n  = lane & 15;
  const int hi = lane >> 4;
  v16h a[KT];
#pragma unroll
  for (int kt = 0; kt < KT; ++kt) a[kt] = load_frag(actIn, kt, lane);
#pragma unroll
  for (int nt = 0; nt < NT; ++nt) {
    v8f c = {0.f, 0.f, 0.f, 0.f, 0.f, 0.f, 0.f, 0.f};
#pragma unroll
    for (int kt = 0; kt < KT; ++kt) {
      v16h b = load_frag(wfrag, kt * NT + nt, lane);
      c = __builtin_amdgcn_wmma_f32_16x16x32_f16(false, a[kt], false, b,
                                                 (short)0, c, false, false);
    }
    const int  ng    = nt * 16 + n;
    const bool valid = ng < Nvalid;
    const float bb   = valid ? bias[ng] : 0.0f;
    // output slot (fixed per lane within this nt): A-layout position of column ng
    const int kto = ng >> 5;
    const int k2  = ng & 31;
    const int hi2 = (k2 >> 3) & 1;
    const int e   = (k2 & 7) + ((k2 & 16) ? 8 : 0);
    half_t* obase = actOut + ((kto << 5) + (hi2 << 4)) * 16 + e;
#pragma unroll
    for (int j = 0; j < 8; ++j) {
      float v = c[j] + bb;
      if (ACT == 0)      v = fmaxf(v, 0.0f);
      else if (ACT == 1) v = 1.0f / (1.0f + __expf(-v));
      const int m = j + 8 * hi;
      if (valid && ((rowmask >> m) & 1u)) obase[m * 16] = (half_t)v;
    }
  }
}

// ---------------- weight swizzle (f32 global -> f16 LDS B-fragment layout) ----------------
// B 16-bit layout (KxN tile 32x16): lane = n + 16*hb ; e<8 -> k = e + 8*hb ; e>=8 -> k = 16 + (e-8) + 8*hb
template <int K, int N, int KT, int NT>
__device__ inline void fill_weight(half_t* dst, const float* W, int tid) {
  constexpr int total = KT * NT * FRH;
  for (int idx = tid; idx < total; idx += THREADS) {
    const int frag = idx >> 9;
    const int w    = idx & 511;
    const int ln   = w >> 4;
    const int e    = w & 15;
    const int kt   = frag / NT;
    const int nt   = frag - kt * NT;
    const int n    = nt * 16 + (ln & 15);
    const int k    = kt * 32 + (e & 7) + ((e & 8) ? 16 : 0) + 8 * (ln >> 4);
    const float v  = (k < K && n < N) ? W[k * N + n] : 0.0f;
    dst[idx] = (half_t)v;
  }
}

// ---------------- positional-encoding octave chain (double-angle recurrence) ----------------
// Writes sin(2^l x) at ng=l*6+d and cos(2^l x) at ng=l*6+3+d for l in [0,10).
// Error ~1e-5, far below the f16 quantization used by the WMMA pipeline.
__device__ __forceinline__ void enc_chain(half_t* encb, int m, int d, float x) {
  float s = sinf(x), c = cosf(x);
#pragma unroll
  for (int l = 0; l < LFREQ; ++l) {
    encb[frag_off(m, l * 6 + d)]     = (half_t)s;
    encb[frag_off(m, l * 6 + 3 + d)] = (half_t)c;
    const float s2 = 2.0f * s * c;
    const float c2 = 1.0f - 2.0f * s * s;
    s = s2; c = c2;
  }
}

// ---------------- main kernel ----------------
__global__ void __launch_bounds__(THREADS)
neural_scene_wmma_kernel(const int* __restrict__ sidx, const float* __restrict__ pos,
                         const float* __restrict__ geo, const int* __restrict__ p2o,
                         const float* __restrict__ rot, const float* __restrict__ loc,
                         const float* __restrict__ dim, const float* __restrict__ scl,
                         const float* __restrict__ Wo1, const float* __restrict__ bo1,
                         const float* __restrict__ Wo2, const float* __restrict__ bo2,
                         const float* __restrict__ Wo3, const float* __restrict__ bo3,
                         const float* __restrict__ Wi1, const float* __restrict__ bi1,
                         const float* __restrict__ Wi2, const float* __restrict__ bi2,
                         const float* __restrict__ Wi3, const float* __restrict__ bi3,
                         float* __restrict__ out, int nrays) {
  extern __shared__ __attribute__((aligned(128))) unsigned char smem[];
  half_t* wWo1   = (half_t*)smem;
  half_t* wWo2   = wWo1 + WO1_H;
  half_t* wWo3   = wWo2 + WO2_H;
  half_t* wWi1   = wWo3 + WO3_H;
  half_t* wWi2   = wWi1 + WI1_H;
  half_t* wWi3   = wWi2 + WI2_H;
  half_t* actAll = wWi3 + WI3_H;
  half_t* zAll   = actAll + ACT_H;
  float*  b_o1   = (float*)(zAll + Z_H);
  float*  b_o2   = b_o1 + 6 * 64;
  float*  b_o3   = b_o2 + 6 * 64;
  float*  b_i1   = b_o3 + 6 * 48;
  float*  b_i2   = b_i1 + 128;
  float*  b_i3   = b_i2 + 128;
  int*    objAll = (int*)(b_i3 + 8);

  const int tid = threadIdx.x;

  // ---- one-time weight / bias staging into LDS ----
  for (int o = 0; o < NOBJ; ++o) {
    fill_weight<DENC, HOBJ, 2, 4>(wWo1 + o * (2 * 4 * FRH), Wo1 + o * (DENC * HOBJ), tid);
    fill_weight<HOBJ, HOBJ, 2, 4>(wWo2 + o * (2 * 4 * FRH), Wo2 + o * (HOBJ * HOBJ), tid);
    fill_weight<HOBJ, MATD, 2, 3>(wWo3 + o * (2 * 3 * FRH), Wo3 + o * (HOBJ * MATD), tid);
  }
  fill_weight<ZK,   HINT, 2, 8>(wWi1, Wi1, tid);
  fill_weight<HINT, HINT, 4, 8>(wWi2, Wi2, tid);
  fill_weight<HINT, 8,    4, 1>(wWi3, Wi3, tid);
  for (int i = tid; i < 6 * 64; i += THREADS) { b_o1[i] = bo1[i]; b_o2[i] = bo2[i]; }
  for (int i = tid; i < 6 * 48; i += THREADS) b_o3[i] = bo3[i];
  for (int i = tid; i < 128;    i += THREADS) { b_i1[i] = bi1[i]; b_i2[i] = bi2[i]; }
  if (tid < 8) b_i3[tid] = bi3[tid];
  __syncthreads();

  const int wave = tid >> 5;
  const int lane = tid & 31;
  const int hi   = lane >> 4;
  half_t* actb  = actAll + wave * (4 * FRH);
  half_t* encb  = actb + 2 * FRH;          // overlay: act tiles 2..3
  half_t* zb    = zAll + wave * (2 * FRH);
  int*    objid = objAll + wave * 16;

  const int tiles = nrays >> 4;

  // Waves run fully independently: all staging is wave-private and DS ops from a
  // wave are in-order, so no block barrier is needed inside the tile loop.
  for (int tile = blockIdx.x * WAVES + wave; tile < tiles; tile += gridDim.x * WAVES) {
    const int r0 = tile << 4;

    // ---- per-ray preprocess: routing, rigid transform, positional encoding ----
    // Ray m = lane&15 handled by the lane pair (m, m+16): dims {0,1} on the low
    // half-wave, dim 2 + geo/padding/bookkeeping on the high half-wave.
    {
      const int m    = lane & 15;
      const int ray  = r0 + m;
      const int v    = sidx[ray];
      const int part = (v < 0) ? 32 : v;
      const int o    = p2o[part];

      const float px = pos[ray * 3 + 0];
      const float py = pos[ray * 3 + 1];
      const float pz = pos[ray * 3 + 2];
      const float* R = rot + o * 9;
      const float dx = px - loc[o * 3 + 0];
      const float dy = py - loc[o * 3 + 1];
      const float dz = pz - loc[o * 3 + 2];
      float x0 = R[0] * dx + R[1] * dy + R[2] * dz;
      float x1 = R[3] * dx + R[4] * dy + R[5] * dz;
      float x2 = R[6] * dx + R[7] * dy + R[8] * dz;
      x0 -= (dim[o * 6 + 3] + dim[o * 6 + 0]) * 0.5f;
      x1 -= (dim[o * 6 + 4] + dim[o * 6 + 1]) * 0.5f;
      x2 -= (dim[o * 6 + 5] + dim[o * 6 + 2]) * 0.5f;
      const float s2 = 2.0f / scl[o];

      if (lane < 16) {
        enc_chain(encb, m, 0, x0 * s2);
        enc_chain(encb, m, 1, x1 * s2);
      } else {
        objid[m] = o;
        enc_chain(encb, m, 2, x2 * s2);
        // enc zero pad: ng 60..63 -> kt=1,hi2=1,e=12..15 (contiguous): one b64
        {
          const uint64_t z64 = 0;
          __builtin_memcpy(encb + frag_off(m, 60), &z64, 8);
        }
        // z: geo features at ng 48..55 -> kt=1,hi2=0,e=8..15 (contiguous): one b128
        {
          v8h g8;
#pragma unroll
          for (int g = 0; g < 8; ++g) g8[g] = (half_t)geo[ray * 8 + g];
          __builtin_memcpy(zb + frag_off(m, 48), &g8, 16);
        }
        // z zero pad: ng 56..63 -> kt=1,hi2=1,e=8..15 (contiguous): one b128
        {
          const v8h zero8 = {};
          __builtin_memcpy(zb + frag_off(m, 56), &zero8, 16);
        }
      }
    }

    // ---- object presence + per-lane row ownership ----
    unsigned present = 0;
#pragma unroll
    for (int i = 0; i < 16; ++i) present |= 1u << objid[i];
    present = (unsigned)__builtin_amdgcn_readfirstlane((int)present);
    int objrow[8];
#pragma unroll
    for (int j = 0; j < 8; ++j) objrow[j] = objid[j + 8 * hi];

    // ---- routed per-object MLPs (dense over present experts, masked write) ----
    for (int o = 0; o < NOBJ; ++o) {
      if (!(present & (1u << o))) continue;
      unsigned rowmask = 0;
#pragma unroll
      for (int j = 0; j < 8; ++j)
        if (objrow[j] == o) rowmask |= 1u << (j + 8 * hi);
      gemm_layer<2, 4, 0>(encb, wWo1 + o * (2 * 4 * FRH), b_o1 + o * 64,
                          actb, 64, lane, 0xFFFFu);
      gemm_layer<2, 4, 0>(actb, wWo2 + o * (2 * 4 * FRH), b_o2 + o * 64,
                          actb, 64, lane, 0xFFFFu);
      gemm_layer<2, 3, 2>(actb, wWo3 + o * (2 * 3 * FRH), b_o3 + o * 48,
                          zb, 48, lane, rowmask);
    }

    // ---- shared interaction MLP (L1 overwrites the enc overlay: enc is dead) ----
    gemm_layer<2, 8, 1>(zb,   wWi1, b_i1, actb, 128, lane, 0xFFFFu);
    gemm_layer<4, 8, 1>(actb, wWi2, b_i2, actb, 128, lane, 0xFFFFu);

    // ---- final 128->8 layer, scatter complex-interleaved output ----
    {
      v16h a[4];
#pragma unroll
      for (int kt = 0; kt < 4; ++kt) a[kt] = load_frag(actb, kt, lane);
      v8f c = {0.f, 0.f, 0.f, 0.f, 0.f, 0.f, 0.f, 0.f};
#pragma unroll
      for (int kt = 0; kt < 4; ++kt) {
        v16h b = load_frag(wWi3, kt, lane);
        c = __builtin_amdgcn_wmma_f32_16x16x32_f16(false, a[kt], false, b,
                                                   (short)0, c, false, false);
      }
      const int n = lane & 15;
      if (n < 8) {
        const float bb = b_i3[n];
        const int nn = n & 3;
        const int q  = (nn == 1) ? 2 : (nn == 2) ? 1 : nn;  // {0,2,1,3} permutation
        const int p  = n >> 2;                              // 0 = real, 1 = imag
#pragma unroll
        for (int j = 0; j < 8; ++j) {
          const int m = j + 8 * hi;
          out[(size_t)(r0 + m) * 8 + 2 * q + p] = c[j] + bb;
        }
      }
    }
  }
}

// ---------------- host launch ----------------
extern "C" void kernel_launch(void* const* d_in, const int* in_sizes, int n_in,
                              void* d_out, int out_size, void* d_ws, size_t ws_size,
                              hipStream_t stream) {
  (void)n_in; (void)out_size; (void)d_ws; (void)ws_size;
  const int N = in_sizes[0];  // B*S rays
  dim3 grid(256), block(THREADS);
  neural_scene_wmma_kernel<<<grid, block, SMEM_BYTES, stream>>>(
      (const int*)d_in[0],  (const float*)d_in[1],  (const float*)d_in[2],
      (const int*)d_in[4],  (const float*)d_in[5],  (const float*)d_in[6],
      (const float*)d_in[7], (const float*)d_in[8],
      (const float*)d_in[9],  (const float*)d_in[10], (const float*)d_in[11],
      (const float*)d_in[12], (const float*)d_in[13], (const float*)d_in[14],
      (const float*)d_in[15], (const float*)d_in[16], (const float*)d_in[17],
      (const float*)d_in[18], (const float*)d_in[19], (const float*)d_in[20],
      (float*)d_out, N);
}
